// TopKSAE_50414326120653
// MI455X (gfx1250) — matrix operand order repaired and checked by reading, not compiled, not run on hardware
//
#include <hip/hip_runtime.h>

typedef __attribute__((ext_vector_type(2)))  float  v2f;
typedef __attribute__((ext_vector_type(8)))  float  v8f;
typedef __attribute__((ext_vector_type(16))) __bf16 v16bf;
typedef __attribute__((ext_vector_type(4)))  int    v4i;

#define D_IN   1024
#define N_FEAT 16384
#define NBATCH 8192
#define KSEL   64

#define BM 128
#define BN 128

// ---------------------------------------------------------------------------
// Async-to-LDS plumbing (CDNA5 GLOBAL_LOAD_ASYNC_TO_LDS_B128, ASYNCcnt)
// Probe-confirmed signature: (v4i as(1)* gsrc, v4i as(3)* lds, Ii off, Ii cpol)
// ---------------------------------------------------------------------------
#if __has_builtin(__builtin_amdgcn_global_load_async_to_lds_b128)
#define ASYNC_BUILTIN 1
#else
#define ASYNC_BUILTIN 0
#endif

__device__ __forceinline__ void async_b128(const void* gsrc, void* ldst) {
#if ASYNC_BUILTIN
  __builtin_amdgcn_global_load_async_to_lds_b128(
      (__attribute__((address_space(1))) v4i*)gsrc,
      (__attribute__((address_space(3))) v4i*)ldst, 0, 0);
#else
  unsigned lds_off = (unsigned)(uintptr_t)ldst;   // flat LDS addr: low 32 bits = LDS offset
  asm volatile("global_load_async_to_lds_b128 %0, %1, off"
               :: "v"(lds_off), "v"(gsrc) : "memory");
#endif
}

__device__ __forceinline__ void wait_async0() {
#if __has_builtin(__builtin_amdgcn_s_wait_asynccnt)
  __builtin_amdgcn_s_wait_asynccnt(0);
#else
  asm volatile("s_wait_asynccnt 0" ::: "memory");
#endif
}

// ---------------------------------------------------------------------------
// bf16 split helpers (round-to-nearest-even)
// ---------------------------------------------------------------------------
__device__ __forceinline__ unsigned bf16_rne(float f) {
  unsigned u = __float_as_uint(f);
  return (u + 0x7fffu + ((u >> 16) & 1u)) >> 16;
}
__device__ __forceinline__ void split1(float a, unsigned short& h, unsigned short& l) {
  unsigned hb = bf16_rne(a);
  float hf = __uint_as_float(hb << 16);
  h = (unsigned short)hb;
  l = (unsigned short)bf16_rne(a - hf);
}

// ---------------------------------------------------------------------------
// Pre-pass: split (x - pre_bias) and W_enc into bf16 hi/lo planes (once).
// ---------------------------------------------------------------------------
__global__ __launch_bounds__(256) void sae_split_x(
    const float* __restrict__ x, const float* __restrict__ pre_bias,
    unsigned short* __restrict__ hi, unsigned short* __restrict__ lo)
{
  size_t i0 = ((size_t)blockIdx.x * 256u + threadIdx.x) * 4u;
  int col = (int)(i0 & (size_t)(D_IN - 1));
  float4 a  = *(const float4*)(x + i0);
  float4 pb = *(const float4*)(pre_bias + col);
  a.x -= pb.x; a.y -= pb.y; a.z -= pb.z; a.w -= pb.w;
  ushort4 h, l;
  split1(a.x, h.x, l.x); split1(a.y, h.y, l.y);
  split1(a.z, h.z, l.z); split1(a.w, h.w, l.w);
  *(ushort4*)(hi + i0) = h;
  *(ushort4*)(lo + i0) = l;
}

__global__ __launch_bounds__(256) void sae_split_w(
    const float* __restrict__ w,
    unsigned short* __restrict__ hi, unsigned short* __restrict__ lo)
{
  size_t i0 = ((size_t)blockIdx.x * 256u + threadIdx.x) * 4u;
  float4 a = *(const float4*)(w + i0);
  ushort4 h, l;
  split1(a.x, h.x, l.x); split1(a.y, h.y, l.y);
  split1(a.z, h.z, l.z); split1(a.w, h.w, l.w);
  *(ushort4*)(hi + i0) = h;
  *(ushort4*)(lo + i0) = l;
}

// ---------------------------------------------------------------------------
// Kernel 1 (fast path): z = Ah*Bh + Ah*Bl + Al*Bh + enc_bias via
// V_WMMA_F32_16X16X32_BF16, double-buffered LDS filled by async-to-LDS.
// ---------------------------------------------------------------------------
#define BKB  64   // K-depth per LDS tile (2 wmma k-steps)
#define LDKB 72   // padded bf16 row stride (=36 dwords): conflict-free + 16B aligned
#define NKT  (D_IN / BKB)   // 16 K-tiles

struct Frag32B { uint4 a, b; };   // 32 bytes -> v16bf

__device__ __forceinline__ v16bf ld_frag(const unsigned short* base, int off0, int off1) {
  Frag32B f;
  f.a = *(const uint4*)(base + off0);
  f.b = *(const uint4*)(base + off1);
  return __builtin_bit_cast(v16bf, f);
}

__global__ __launch_bounds__(256) void sae_enc_gemm_bf16x3(
    const unsigned short* __restrict__ Ah, const unsigned short* __restrict__ Al,
    const unsigned short* __restrict__ Bh, const unsigned short* __restrict__ Bl,
    const float* __restrict__ enc_bias, float* __restrict__ z)
{
  __shared__ unsigned short sAh[2][BM * LDKB], sAl[2][BM * LDKB];
  __shared__ unsigned short sBh[2][BN * LDKB], sBl[2][BN * LDKB];

  const int tid = threadIdx.x;
  const int bm0 = blockIdx.y * BM;
  const int bn0 = blockIdx.x * BN;

  const int lane   = tid & 31;
  const int wave   = tid >> 5;
  const int lane16 = lane & 15;
  const int khalf  = lane >> 4;
  const int wm = (wave & 1) * 64;
  const int wn = (wave >> 1) * 32;

  const int lrow  = tid >> 3;        // 0..31
  const int lcol8 = (tid & 7) * 8;   // bf16 column, 16B chunks

  v8f acc[4][2];
  const v8f vzero = {0.f,0.f,0.f,0.f,0.f,0.f,0.f,0.f};
  #pragma unroll
  for (int mt = 0; mt < 4; ++mt)
    #pragma unroll
    for (int nt = 0; nt < 2; ++nt) acc[mt][nt] = vzero;

  // -- issue one K-tile's async fills into buffer `buf`
  auto issue_tile = [&](int buf, int k0) {
    #pragma unroll
    for (int i = 0; i < 4; ++i) {
      int r = lrow + 32 * i;
      size_t ga = (size_t)(bm0 + r) * D_IN + k0 + lcol8;
      size_t gb = (size_t)(bn0 + r) * D_IN + k0 + lcol8;
      int lo = r * LDKB + lcol8;
      async_b128(Ah + ga, &sAh[buf][lo]);
      async_b128(Al + ga, &sAl[buf][lo]);
      async_b128(Bh + gb, &sBh[buf][lo]);
      async_b128(Bl + gb, &sBl[buf][lo]);
    }
  };

  issue_tile(0, 0);
  wait_async0();
  __syncthreads();

  for (int t = 0; t < NKT; ++t) {
    const int cur = t & 1;
    if (t + 1 < NKT) issue_tile(cur ^ 1, (t + 1) * BKB);

    const unsigned short* pAh = sAh[cur];
    const unsigned short* pAl = sAl[cur];
    const unsigned short* pBh = sBh[cur];
    const unsigned short* pBl = sBl[cur];

    #pragma unroll
    for (int kk = 0; kk < BKB; kk += 32) {
      // A frag (16-bit A 16x32): lane<16 -> K[kk+0..7],[kk+16..23]; lane>=16 -> +8
      v16bf afh[4], afl[4], bfh[2], bfl[2];
      #pragma unroll
      for (int mt = 0; mt < 4; ++mt) {
        int row = (wm + mt * 16 + lane16) * LDKB;
        int o0 = row + kk + 8 * khalf;
        int o1 = row + kk + 16 + 8 * khalf;
        afh[mt] = ld_frag(pAh, o0, o1);
        afl[mt] = ld_frag(pAl, o0, o1);
      }
      // B frag (16-bit B 32x16): lane<16 -> K[kk+0..7],[kk+16..23]; lane>=16 -> +8
      #pragma unroll
      for (int nt = 0; nt < 2; ++nt) {
        int row = (wn + nt * 16 + lane16) * LDKB;
        int o0 = row + kk + 8 * khalf;
        bfh[nt] = ld_frag(pBh, o0, o0 + 16);
        bfl[nt] = ld_frag(pBl, o0, o0 + 16);
      }
      #pragma unroll
      for (int mt = 0; mt < 4; ++mt)
        #pragma unroll
        for (int nt = 0; nt < 2; ++nt) {
          acc[mt][nt] = __builtin_amdgcn_wmma_f32_16x16x32_bf16(
              false, afh[mt], false, bfh[nt], (short)0, acc[mt][nt], false, false);
          acc[mt][nt] = __builtin_amdgcn_wmma_f32_16x16x32_bf16(
              false, afh[mt], false, bfl[nt], (short)0, acc[mt][nt], false, false);
          acc[mt][nt] = __builtin_amdgcn_wmma_f32_16x16x32_bf16(
              false, afl[mt], false, bfh[nt], (short)0, acc[mt][nt], false, false);
        }
    }

    wait_async0();     // this wave's fills for tile t+1 are in LDS
    __syncthreads();   // everyone done reading buf `cur` + all fills visible
  }

  #pragma unroll
  for (int nt = 0; nt < 2; ++nt) {
    int col = bn0 + wn + nt * 16 + lane16;
    float eb = enc_bias[col];
    #pragma unroll
    for (int mt = 0; mt < 4; ++mt)
      #pragma unroll
      for (int r = 0; r < 8; ++r) {
        int rowg = bm0 + wm + mt * 16 + khalf * 8 + r;
        z[(size_t)rowg * N_FEAT + col] = acc[mt][nt][r] + eb;
      }
  }
}

// ---------------------------------------------------------------------------
// Kernel 1 (fallback, full fp32): V_WMMA_F32_16X16X4_F32 — used when d_ws is
// too small to hold the bf16 split planes.
// ---------------------------------------------------------------------------
#define BK  32
#define LDK 36

__global__ __launch_bounds__(256) void sae_enc_gemm_f32(
    const float* __restrict__ x, const float* __restrict__ Wenc,
    const float* __restrict__ enc_bias, const float* __restrict__ pre_bias,
    float* __restrict__ z)
{
  __shared__ float As[BM * LDK];
  __shared__ float Bs[BN * LDK];

  const int tid = threadIdx.x;
  const int bm0 = blockIdx.y * BM;
  const int bn0 = blockIdx.x * BN;

  const int lane   = tid & 31;
  const int wave   = tid >> 5;
  const int lane16 = lane & 15;
  const int khalf  = lane >> 4;
  const int wm = (wave & 1) * 64;
  const int wn = (wave >> 1) * 32;

  const int lrow = tid >> 3;
  const int lcol = (tid & 7) * 4;

  v8f acc[4][2];
  const v8f vzero = {0.f,0.f,0.f,0.f,0.f,0.f,0.f,0.f};
  #pragma unroll
  for (int mt = 0; mt < 4; ++mt)
    #pragma unroll
    for (int nt = 0; nt < 2; ++nt) acc[mt][nt] = vzero;

  for (int k0 = 0; k0 < D_IN; k0 += BK) {
    __syncthreads();
    float4 pb = *(const float4*)(pre_bias + k0 + lcol);
    #pragma unroll
    for (int i = 0; i < 4; ++i) {
      int r = lrow + 32 * i;
      float4 a = *(const float4*)(x + (size_t)(bm0 + r) * D_IN + k0 + lcol);
      a.x -= pb.x; a.y -= pb.y; a.z -= pb.z; a.w -= pb.w;
      *(float4*)&As[r * LDK + lcol] = a;
      *(float4*)&Bs[r * LDK + lcol] = *(const float4*)(Wenc + (size_t)(bn0 + r) * D_IN + k0 + lcol);
    }
    __syncthreads();

    #pragma unroll
    for (int kk = 0; kk < BK; kk += 4) {
      v2f af[4], bf[2];
      #pragma unroll
      for (int mt = 0; mt < 4; ++mt)
        af[mt] = *(const v2f*)&As[(wm + mt * 16 + lane16) * LDK + kk + 2 * khalf];
      #pragma unroll
      for (int nt = 0; nt < 2; ++nt)
        bf[nt] = *(const v2f*)&Bs[(wn + nt * 16 + lane16) * LDK + kk + 2 * khalf];
      #pragma unroll
      for (int mt = 0; mt < 4; ++mt)
        #pragma unroll
        for (int nt = 0; nt < 2; ++nt)
          acc[mt][nt] = __builtin_amdgcn_wmma_f32_16x16x4_f32(
              false, af[mt], false, bf[nt], (short)0, acc[mt][nt], false, false);
    }
  }

  #pragma unroll
  for (int nt = 0; nt < 2; ++nt) {
    int col = bn0 + wn + nt * 16 + lane16;
    float eb = enc_bias[col];
    #pragma unroll
    for (int mt = 0; mt < 4; ++mt)
      #pragma unroll
      for (int r = 0; r < 8; ++r) {
        int rowg = bm0 + wm + mt * 16 + khalf * 8 + r;
        z[(size_t)rowg * N_FEAT + col] = acc[mt][nt][r] + eb;
      }
  }
}

// ---------------------------------------------------------------------------
// Kernel 2: per-row abs-top-64 via 3-pass radix select on abs float bits,
// sparsify the z row in place, emit indices sorted by descending |v|.
// ---------------------------------------------------------------------------
__global__ __launch_bounds__(256) void sae_topk(
    float* __restrict__ z, float* __restrict__ idx_out)
{
  __shared__ unsigned hist[2048];
  __shared__ unsigned partial[256];
  __shared__ unsigned bcast[2];
  __shared__ float    sel_val[KSEL];
  __shared__ int      sel_idx[KSEL];
  __shared__ unsigned cnt[2];

  const int tid = threadIdx.x;
  float* zrow = z + (size_t)blockIdx.x * N_FEAT;

  unsigned pref = 0;
  int prefBits = 0;
  unsigned Krem = KSEL;

  for (int pass = 0; pass < 3; ++pass) {
    const int width = (pass == 0) ? 11 : 10;     // 11+10+10 = 31 bits
    const int nbins = 1 << width;
    const int shift = 31 - prefBits - width;

    for (int i = tid; i < nbins; i += 256) hist[i] = 0;
    __syncthreads();

    for (int i = tid; i < N_FEAT; i += 256) {
      unsigned u = __float_as_uint(zrow[i]) & 0x7fffffffu;
      if ((u >> (31 - prefBits)) == pref)
        atomicAdd(&hist[(u >> shift) & (unsigned)(nbins - 1)], 1u);
    }
    __syncthreads();

    const int cs = nbins >> 8;
    unsigned s = 0;
    for (int j = 0; j < cs; ++j) s += hist[tid * cs + j];
    partial[tid] = s;
    __syncthreads();

    if (tid == 0) {
      unsigned cum = 0;
      int chunk = 255;
      for (; chunk > 0; --chunk) {
        if (cum + partial[chunk] >= Krem) break;
        cum += partial[chunk];
      }
      int d = chunk * cs + cs - 1;
      for (; d > chunk * cs; --d) {
        if (cum + hist[d] >= Krem) break;
        cum += hist[d];
      }
      bcast[0] = (unsigned)d;
      bcast[1] = Krem - cum;
    }
    __syncthreads();
    pref = (pref << width) | bcast[0];
    Krem = bcast[1];
    prefBits += width;
    __syncthreads();
  }

  if (tid < 2) cnt[tid] = 0;
  __syncthreads();
  const unsigned T = pref;
  const unsigned needEq = Krem;
  const unsigned nGt = KSEL - needEq;

  for (int i = tid; i < N_FEAT; i += 256) {
    float v = zrow[i];
    unsigned u = __float_as_uint(v) & 0x7fffffffu;
    if (u > T) {
      unsigned p = atomicAdd(&cnt[0], 1u);
      sel_val[p] = v; sel_idx[p] = i;
    } else if (u == T) {
      unsigned p = atomicAdd(&cnt[1], 1u);
      if (p < needEq) { sel_val[nGt + p] = v; sel_idx[nGt + p] = i; }
    }
  }
  __syncthreads();

  for (int phase = 0; phase < KSEL; ++phase) {
    int j = 2 * tid + (phase & 1);
    if (j + 1 < KSEL) {
      float va = sel_val[j], vb = sel_val[j + 1];
      unsigned ua = __float_as_uint(va) & 0x7fffffffu;
      unsigned ub = __float_as_uint(vb) & 0x7fffffffu;
      if (ub > ua) {
        int ia = sel_idx[j], ib = sel_idx[j + 1];
        sel_val[j] = vb; sel_val[j + 1] = va;
        sel_idx[j] = ib; sel_idx[j + 1] = ia;
      }
    }
    __syncthreads();
  }

  float4 z4 = {0.f, 0.f, 0.f, 0.f};
  for (int i = tid; i < N_FEAT / 4; i += 256) ((float4*)zrow)[i] = z4;
  __syncthreads();
  if (tid < KSEL) {
    zrow[sel_idx[tid]] = sel_val[tid];
    idx_out[(size_t)blockIdx.x * KSEL + tid] = (float)sel_idx[tid];
  }
}

// ---------------------------------------------------------------------------
// Kernel 3: decode. W_dec[d,f] == W_enc[f,d] (tied), gather contiguous
// L2-resident rows of W_enc.
// ---------------------------------------------------------------------------
__global__ __launch_bounds__(256) void sae_decode(
    const float* __restrict__ Wenc, const float* __restrict__ pre_bias,
    const float* __restrict__ dec_bias, const float* __restrict__ z,
    const float* __restrict__ idx_f, float* __restrict__ xhat)
{
  __shared__ float sval[KSEL];
  __shared__ int   sidx[KSEL];
  const int tid = threadIdx.x;
  const size_t b = blockIdx.x;

  if (tid < KSEL) {
    int f = (int)idx_f[b * KSEL + tid];
    sidx[tid] = f;
    sval[tid] = z[b * (size_t)N_FEAT + f];
  }
  __syncthreads();

  float acc[4];
  #pragma unroll
  for (int j = 0; j < 4; ++j) {
    int d = tid + 256 * j;
    acc[j] = dec_bias[d] + pre_bias[d];
  }
  for (int k = 0; k < KSEL; ++k) {
    float v = sval[k];
    const float* col = Wenc + (size_t)sidx[k] * D_IN;
    #pragma unroll
    for (int j = 0; j < 4; ++j)
      acc[j] += v * col[tid + 256 * j];
  }
  #pragma unroll
  for (int j = 0; j < 4; ++j)
    xhat[b * D_IN + tid + 256 * j] = acc[j];
}

// ---------------------------------------------------------------------------
extern "C" void kernel_launch(void* const* d_in, const int* in_sizes, int n_in,
                              void* d_out, int out_size, void* d_ws, size_t ws_size,
                              hipStream_t stream) {
  (void)in_sizes; (void)n_in; (void)out_size;
  const float* x        = (const float*)d_in[0];
  const float* Wenc     = (const float*)d_in[1];
  const float* enc_bias = (const float*)d_in[2];
  const float* pre_bias = (const float*)d_in[3];
  // d_in[4] = W_dec (unused: W_dec == W_enc^T exactly, tied init)
  const float* dec_bias = (const float*)d_in[5];

  float* out    = (float*)d_out;
  float* xhat   = out;                                // [8192,1024]
  float* zout   = out + (size_t)NBATCH * D_IN;        // [8192,16384]
  float* idxout = zout + (size_t)NBATCH * N_FEAT;     // [8192,64] as float

  dim3 g1(N_FEAT / BN, NBATCH / BM);                  // 128 x 64 workgroups

  const size_t nA = (size_t)NBATCH * D_IN;            // 8,388,608
  const size_t nB = (size_t)N_FEAT * D_IN;            // 16,777,216
  const size_t wsNeed = 2 * (nA + nB) * sizeof(unsigned short);  // 96 MB

  if (ws_size >= wsNeed) {
    unsigned short* Ah = (unsigned short*)d_ws;
    unsigned short* Al = Ah + nA;
    unsigned short* Bh = Al + nA;
    unsigned short* Bl = Bh + nB;
    sae_split_x<<<(unsigned)(nA / 1024), 256, 0, stream>>>(x, pre_bias, Ah, Al);
    sae_split_w<<<(unsigned)(nB / 1024), 256, 0, stream>>>(Wenc, Bh, Bl);
    sae_enc_gemm_bf16x3<<<g1, 256, 0, stream>>>(Ah, Al, Bh, Bl, enc_bias, zout);
  } else {
    sae_enc_gemm_f32<<<g1, 256, 0, stream>>>(x, Wenc, enc_bias, pre_bias, zout);
  }

  sae_topk<<<NBATCH, 256, 0, stream>>>(zout, idxout);
  sae_decode<<<NBATCH, 256, 0, stream>>>(Wenc, pre_bias, dec_bias, zout, idxout, xhat);
}
